// MultiheadExponentialTimeDecay_81329500717659
// MI455X (gfx1250) — matrix operand order repaired and checked by reading, not compiled, not run on hardware
//
#include <hip/hip_runtime.h>
#include <hip/hip_bf16.h>

// ---------------------------------------------------------------------------
// MultiheadExponentialTimeDecay for MI455X (gfx1250, wave32, WMMA)
//
// reference = LN -> GEMM(w_in) -> causal EMA per (b,head,dh) -> GEMM(w_out)
// (the FFT in the reference is just a causal exponential filter:
//  y_t = (1-a) y_{t-1} + a x_t)
// ---------------------------------------------------------------------------

#define B_  4
#define N_  4096
#define D_  1024
#define H_  16
#define INNER_ 1024
#define M_  (B_ * N_)      // 16384 rows
#define SEG 256
#define NSEG (N_ / SEG)    // 16

typedef _Float16 v16h __attribute__((ext_vector_type(16)));
typedef _Float16 v8h  __attribute__((ext_vector_type(8)));
typedef _Float16 v4h  __attribute__((ext_vector_type(4)));
typedef float    v8f  __attribute__((ext_vector_type(8)));

union FragU { v16h v; v8h h[2]; };

// ---------------------------------------------------------------------------
// 1) LayerNorm over D=1024, emit f16 activations
// ---------------------------------------------------------------------------
__global__ __launch_bounds__(256)
void ln_f16_kernel(const float* __restrict__ x,
                   const float* __restrict__ gamma,
                   const float* __restrict__ beta,
                   _Float16* __restrict__ out)
{
    const int row = blockIdx.x;            // 0..M-1
    const int t   = threadIdx.x;           // 0..255, 4 elems each
    const float4 v = ((const float4*)(x + (size_t)row * D_))[t];

    __shared__ float sh1[256], sh2[256];
    sh1[t] = v.x + v.y + v.z + v.w;
    sh2[t] = v.x * v.x + v.y * v.y + v.z * v.z + v.w * v.w;
    __syncthreads();
    for (int off = 128; off > 0; off >>= 1) {
        if (t < off) { sh1[t] += sh1[t + off]; sh2[t] += sh2[t + off]; }
        __syncthreads();
    }
    const float mu  = sh1[0] * (1.0f / (float)D_);
    const float var = sh2[0] * (1.0f / (float)D_) - mu * mu;
    const float inv = rsqrtf(var + 1e-5f);

    const float4 g  = ((const float4*)gamma)[t];
    const float4 bb = ((const float4*)beta)[t];
    v4h o;
    o[0] = (_Float16)((v.x - mu) * inv * g.x + bb.x);
    o[1] = (_Float16)((v.y - mu) * inv * g.y + bb.y);
    o[2] = (_Float16)((v.z - mu) * inv * g.z + bb.z);
    o[3] = (_Float16)((v.w - mu) * inv * g.w + bb.w);
    *(v4h*)(out + (size_t)row * D_ + t * 4) = o;
}

// ---------------------------------------------------------------------------
// 2) f32 -> f16 weight conversion
// ---------------------------------------------------------------------------
__global__ __launch_bounds__(256)
void f32_to_f16_kernel(const float* __restrict__ in,
                       _Float16* __restrict__ out, int n4)
{
    const int i = blockIdx.x * 256 + threadIdx.x;
    if (i < n4) {
        const float4 v = ((const float4*)in)[i];
        v4h o;
        o[0] = (_Float16)v.x; o[1] = (_Float16)v.y;
        o[2] = (_Float16)v.z; o[3] = (_Float16)v.w;
        *(v4h*)(out + (size_t)i * 4) = o;
    }
}

// ---------------------------------------------------------------------------
// 3) WMMA f16 GEMM:  C[m,j] = sum_k A[m,k] * W[j,k]
//    A: M x K row-major f16, W: Nout x K row-major f16, C: M x Nout f32.
//    Block tile 128x128, BK=32 (one v_wmma_f32_16x16x32_f16 per K-panel),
//    8 waves of 64x32 each (4x2 16x16 accumulators).
//    Double-buffered LDS fed by GLOBAL_LOAD_ASYNC_TO_LDS_B128 (ASYNCcnt),
//    software-pipelined one K-panel ahead of the WMMA stream.
// ---------------------------------------------------------------------------
#define BK   32
#define LDSK 40   // padded LDS row stride in halfs (80B, 16B aligned, bank-spread)

__global__ __launch_bounds__(256)
void gemm_f16_wmma(const _Float16* __restrict__ A,
                   const _Float16* __restrict__ W,
                   float* __restrict__ C,
                   int M, int Nout, int K)
{
    __shared__ _Float16 As[2][128 * LDSK];
    __shared__ _Float16 Ws[2][128 * LDSK];

    const int tid     = threadIdx.x;
    const int lane    = tid & 31;
    const int wv      = tid >> 5;            // 0..7
    const int waveM   = (wv >> 2) * 64;      // 0 / 64
    const int waveN   = (wv & 3) * 32;       // 0..96
    const int laneRow = lane & 15;
    const int laneHi  = lane >> 4;           // 0 or 1

    const int blockM = blockIdx.y * 128;
    const int blockN = blockIdx.x * 128;

    // this thread's two 16B staging slots (512 slots cover a 128x32 panel)
    const int row0 = tid >> 2;               // slot 0: rows 0..63
    const int ch0  = (tid & 3) * 8;
    const int row1 = (tid + 256) >> 2;       // slot 1: rows 64..127
    const int ch1  = ch0;

    // issue one K-panel of async global->LDS copies (4 instrs per thread/wave)
    auto stage_async = [&](int k0, int buf) {
        {
            const void* ga = A + (size_t)(blockM + row0) * K + k0 + ch0;
            const void* gw = W + (size_t)(blockN + row0) * K + k0 + ch0;
            unsigned la = (unsigned)(size_t)&As[buf][row0 * LDSK + ch0];
            unsigned lw = (unsigned)(size_t)&Ws[buf][row0 * LDSK + ch0];
            asm volatile("global_load_async_to_lds_b128 %0, %1, off"
                         :: "v"(la), "v"(ga) : "memory");
            asm volatile("global_load_async_to_lds_b128 %0, %1, off"
                         :: "v"(lw), "v"(gw) : "memory");
        }
        {
            const void* ga = A + (size_t)(blockM + row1) * K + k0 + ch1;
            const void* gw = W + (size_t)(blockN + row1) * K + k0 + ch1;
            unsigned la = (unsigned)(size_t)&As[buf][row1 * LDSK + ch1];
            unsigned lw = (unsigned)(size_t)&Ws[buf][row1 * LDSK + ch1];
            asm volatile("global_load_async_to_lds_b128 %0, %1, off"
                         :: "v"(la), "v"(ga) : "memory");
            asm volatile("global_load_async_to_lds_b128 %0, %1, off"
                         :: "v"(lw), "v"(gw) : "memory");
        }
    };

    v8f acc[4][2];
#pragma unroll
    for (int i = 0; i < 4; ++i)
#pragma unroll
        for (int j = 0; j < 2; ++j) acc[i][j] = {};

    stage_async(0, 0);                       // prologue: panel 0 -> buf 0
    int buf = 0;

    for (int k0 = 0; k0 < K; k0 += BK) {
        // pipeline: start panel k+1 (clamped; last iter re-fetches harmlessly
        // so the ASYNCcnt wait immediate is always exactly 4)
        int kn = k0 + BK;
        if (kn >= K) kn = K - BK;
        stage_async(kn, buf ^ 1);

        // wait for the CURRENT panel's 4 loads (the 4 just issued stay in flight)
        asm volatile("s_wait_asynccnt 0x4" ::: "memory");
        __syncthreads();

        // B fragments: lane(0-15)=col, K 0..7 & 16..23 ; lane(16-31)=col, K 8..15 & 24..31
        v16h bfrag[2];
#pragma unroll
        for (int j = 0; j < 2; ++j) {
            const int col = waveN + j * 16 + laneRow;
            FragU f;
            f.h[0] = *(const v8h*)(&Ws[buf][col * LDSK + laneHi * 8]);
            f.h[1] = *(const v8h*)(&Ws[buf][col * LDSK + laneHi * 8 + 16]);
            bfrag[j] = f.v;
        }
        // A fragments: lane(0-15): K 0..7 & 16..23 ; lane(16-31): K 8..15 & 24..31
        v16h afrag[4];
#pragma unroll
        for (int i = 0; i < 4; ++i) {
            const int row = waveM + i * 16 + laneRow;
            FragU f;
            f.h[0] = *(const v8h*)(&As[buf][row * LDSK + laneHi * 8]);
            f.h[1] = *(const v8h*)(&As[buf][row * LDSK + laneHi * 8 + 16]);
            afrag[i] = f.v;
        }

#pragma unroll
        for (int i = 0; i < 4; ++i)
#pragma unroll
            for (int j = 0; j < 2; ++j)
                acc[i][j] = __builtin_amdgcn_wmma_f32_16x16x32_f16(
                    false, afrag[i], false, bfrag[j],
                    (short)0, acc[i][j], false, false);

        // everyone done reading buf before next iteration overwrites it
        __syncthreads();
        buf ^= 1;
    }

    // C/D layout: VGPR v, lanes 0-15 -> M = tile+v, lanes 16-31 -> M = tile+8+v
#pragma unroll
    for (int i = 0; i < 4; ++i)
#pragma unroll
        for (int j = 0; j < 2; ++j) {
            const int r0 = blockM + waveM + i * 16 + laneHi * 8;
            const int col = blockN + waveN + j * 16 + laneRow;
            float* cp = C + (size_t)r0 * Nout + col;
#pragma unroll
            for (int v = 0; v < 8; ++v)
                cp[(size_t)v * Nout] = acc[i][j][v];
        }
}

// ---------------------------------------------------------------------------
// 4) EMA scan, pass A: per-segment local scan, in place. One lane per
//    (b, e, seg); lanes contiguous in e -> coalesced 4KB row reads.
// ---------------------------------------------------------------------------
__global__ __launch_bounds__(1024)
void ema_scan_seg_kernel(float* __restrict__ xi,
                         const float* __restrict__ alpha,
                         float* __restrict__ carryL)
{
    const int e   = threadIdx.x;             // 0..1023
    const int b   = blockIdx.x / NSEG;
    const int seg = blockIdx.x % NSEG;
    const int h   = e >> 6;
    const float a = 1.0f / (1.0f + __expf(-alpha[h]));
    const float r = 1.0f - a;

    size_t idx = ((size_t)b * N_ + (size_t)seg * SEG) * INNER_ + e;
    float y = 0.0f;
    for (int t = 0; t < SEG; ++t) {
        y = fmaf(r, y, a * xi[idx]);
        xi[idx] = y;
        idx += INNER_;
    }
    carryL[((size_t)b * INNER_ + e) * NSEG + seg] = y;
}

// 4b) pass B: combine segment carries per channel. C_s = L_s + r^SEG * C_{s-1};
//     carryP[s] = C_{s-1} (incoming EMA state for segment s).
__global__ __launch_bounds__(256)
void ema_carry_kernel(const float* __restrict__ carryL,
                      const float* __restrict__ alpha,
                      float* __restrict__ carryP)
{
    const int ch = blockIdx.x * 256 + threadIdx.x;   // 0..4095 = b*1024+e
    if (ch >= B_ * INNER_) return;
    const int e = ch & (INNER_ - 1);
    const int h = e >> 6;
    const float a  = 1.0f / (1.0f + __expf(-alpha[h]));
    const float r  = 1.0f - a;
    const float rS = __powf(r, (float)SEG);
    float c = 0.0f;
    for (int s = 0; s < NSEG; ++s) {
        carryP[(size_t)ch * NSEG + s] = c;
        c = carryL[(size_t)ch * NSEG + s] + rS * c;
    }
}

// 4c) pass C: y = y_local + r^(t_local+1) * carry, convert to f16 for GEMM-2.
__global__ __launch_bounds__(256)
void ema_fix_f16_kernel(const float* __restrict__ xi,
                        const float* __restrict__ carryP,
                        const float* __restrict__ alpha,
                        _Float16* __restrict__ yh)
{
    const size_t tg = (size_t)blockIdx.x * 256 + threadIdx.x;  // float4 index
    const int e4 = (int)(tg & 255);
    const size_t bn = tg >> 8;                 // 0..16383
    const int n = (int)(bn & (N_ - 1));
    const int b = (int)(bn >> 12);
    const int e = e4 * 4;
    const int h = e >> 6;
    const float a = 1.0f / (1.0f + __expf(-alpha[h]));
    const float r = 1.0f - a;
    const int seg = n >> 8;                    // n / SEG
    const int tl  = n & (SEG - 1);
    const float corr = __powf(r, (float)(tl + 1));

    const float4 yv = ((const float4*)xi)[tg];
    const size_t cb = ((size_t)b * INNER_ + e) * NSEG + seg;
    const float p0 = carryP[cb];
    const float p1 = carryP[cb + NSEG];
    const float p2 = carryP[cb + 2 * NSEG];
    const float p3 = carryP[cb + 3 * NSEG];

    v4h o;
    o[0] = (_Float16)(yv.x + corr * p0);
    o[1] = (_Float16)(yv.y + corr * p1);
    o[2] = (_Float16)(yv.z + corr * p2);
    o[3] = (_Float16)(yv.w + corr * p3);
    *(v4h*)(yh + tg * 4) = o;
}

// ---------------------------------------------------------------------------
// Launcher
// ---------------------------------------------------------------------------
extern "C" void kernel_launch(void* const* d_in, const int* in_sizes, int n_in,
                              void* d_out, int out_size, void* d_ws, size_t ws_size,
                              hipStream_t stream)
{
    const float* x     = (const float*)d_in[0];
    const float* gamma = (const float*)d_in[1];
    const float* beta  = (const float*)d_in[2];
    const float* alpha = (const float*)d_in[3];
    const float* w_in  = (const float*)d_in[4];
    const float* w_out = (const float*)d_in[5];
    float* out = (float*)d_out;

    char* ws = (char*)d_ws;
    const size_t SZ_XNH = (size_t)M_ * D_ * 2;        // 32 MB  f16 LN activations
    const size_t SZ_WH  = (size_t)INNER_ * D_ * 2;    //  2 MB  f16 weights (each)
    const size_t SZ_XI  = (size_t)M_ * INNER_ * 4;    // 64 MB  f32 xi / y_local
    const size_t SZ_CL  = (size_t)B_ * INNER_ * NSEG * 4;  // 256 KB
    _Float16* xnh    = (_Float16*)(ws);
    _Float16* w_inh  = (_Float16*)(ws + SZ_XNH);
    _Float16* w_outh = (_Float16*)(ws + SZ_XNH + SZ_WH);
    float*    xi     = (float*)   (ws + SZ_XNH + 2 * SZ_WH);
    float*    carryL = (float*)   (ws + SZ_XNH + 2 * SZ_WH + SZ_XI);
    float*    carryP = (float*)   (ws + SZ_XNH + 2 * SZ_WH + SZ_XI + SZ_CL);
    _Float16* yh     = (_Float16*)(ws + SZ_XNH + 2 * SZ_WH + SZ_XI + 2 * SZ_CL);

    // 1) LayerNorm -> f16
    ln_f16_kernel<<<M_, 256, 0, stream>>>(x, gamma, beta, xnh);

    // 2) weights -> f16
    f32_to_f16_kernel<<<(INNER_ * D_ / 4 + 255) / 256, 256, 0, stream>>>(
        w_in, w_inh, INNER_ * D_ / 4);
    f32_to_f16_kernel<<<(D_ * INNER_ / 4 + 255) / 256, 256, 0, stream>>>(
        w_out, w_outh, D_ * INNER_ / 4);

    // 3) xi = xn @ w_in^T   (M x INNER, K = D)
    gemm_f16_wmma<<<dim3(INNER_ / 128, M_ / 128), 256, 0, stream>>>(
        xnh, w_inh, xi, M_, INNER_, D_);

    // 4) causal EMA scan (segmented), producing f16 activations for GEMM-2
    ema_scan_seg_kernel<<<B_ * NSEG, 1024, 0, stream>>>(xi, alpha, carryL);
    ema_carry_kernel<<<(B_ * INNER_ + 255) / 256, 256, 0, stream>>>(
        carryL, alpha, carryP);
    ema_fix_f16_kernel<<<(int)((size_t)M_ * INNER_ / 4 / 256), 256, 0, stream>>>(
        xi, carryP, alpha, yh);

    // 5) out = y @ w_out^T  (M x D, K = INNER)
    gemm_f16_wmma<<<dim3(D_ / 128, M_ / 128), 256, 0, stream>>>(
        yh, w_outh, out, M_, D_, INNER_);

    (void)in_sizes; (void)n_in; (void)out_size; (void)ws_size;
}
